// LuongAttention_45165876084978
// MI455X (gfx1250) — matrix operand order repaired and checked by reading, not compiled, not run on hardware
//
#include <hip/hip_runtime.h>

typedef __attribute__((ext_vector_type(16))) __bf16 v16bf;
typedef __attribute__((ext_vector_type(8)))  __bf16 v8bf;
typedef __attribute__((ext_vector_type(8)))  float  v8f;
typedef __attribute__((ext_vector_type(4)))  float  v4f;

#define B_ 4
#define S_ 4096
#define Q_ 4096
#define D_ 1024

// ---------------------------------------------------------------------------
// Pack fp32 -> bf16 (flat)
// ---------------------------------------------------------------------------
__global__ __launch_bounds__(256)
void k_pack_bf16(const float* __restrict__ in, __bf16* __restrict__ out, int n) {
    int i = (blockIdx.x * 256 + threadIdx.x) * 4;
    if (i + 3 < n) {
        v4f v = *(const v4f*)(in + i);
        out[i + 0] = (__bf16)v[0];
        out[i + 1] = (__bf16)v[1];
        out[i + 2] = (__bf16)v[2];
        out[i + 3] = (__bf16)v[3];
    }
}

// ---------------------------------------------------------------------------
// value [B,S,D] fp32 -> valueT [B,D,S] bf16, LDS 32x33 tile transpose
// ---------------------------------------------------------------------------
__global__ __launch_bounds__(256)
void k_pack_valueT(const float* __restrict__ v, __bf16* __restrict__ vt) {
    __shared__ __bf16 tile[32][33];
    const int dt = D_ / 32;            // tiles along d
    const int st = S_ / 32;            // tiles along s
    int bid = blockIdx.x;
    int b   = bid / (st * dt);
    int r   = bid % (st * dt);
    int s0  = (r / dt) * 32;
    int d0  = (r % dt) * 32;
    int tx  = threadIdx.x & 31;
    int ty  = threadIdx.x >> 5;

    const float* src = v + (size_t)b * S_ * D_;
    #pragma unroll
    for (int y = ty; y < 32; y += 8)
        tile[y][tx] = (__bf16)src[(size_t)(s0 + y) * D_ + d0 + tx];
    __syncthreads();
    __bf16* dst = vt + (size_t)b * D_ * S_;
    #pragma unroll
    for (int y = ty; y < 32; y += 8)
        dst[(size_t)(d0 + y) * S_ + s0 + tx] = tile[tx][y];
}

// ---------------------------------------------------------------------------
// GEMM 1: kW[b,s,k] = sum_d key[b,s,d] * W[k,d]   (A fp32, B bf16, C bf16)
// One wave per 64x64 tile of [S_, D_] output.
// ---------------------------------------------------------------------------
__global__ __launch_bounds__(32)
void k_gemm_kW(const float* __restrict__ key, const __bf16* __restrict__ Wb,
               __bf16* __restrict__ kWb) {
    const int lane = threadIdx.x;
    const int ntiles = D_ / 64, mtiles = S_ / 64;
    int bid = blockIdx.x;
    int b  = bid / (mtiles * ntiles);
    int r  = bid % (mtiles * ntiles);
    int m0 = (r / ntiles) * 64;
    int n0 = (r % ntiles) * 64;

    const float* Ab = key + (size_t)b * S_ * D_;
    __bf16*      Cb = kWb + (size_t)b * S_ * D_;

    const int lm   = lane & 15;
    const int hi   = lane >> 4;       // 0 or 1
    const int kb8  = hi * 8;
    const int kb16 = hi * 16;

    v8f acc[4][4];
    #pragma unroll
    for (int i = 0; i < 4; i++)
        #pragma unroll
        for (int j = 0; j < 4; j++) acc[i][j] = (v8f)0.0f;

    for (int k0 = 0; k0 < D_; k0 += 32) {
        v16bf bf[4];
        #pragma unroll
        for (int j = 0; j < 4; j++) {
            int col = n0 + j * 16 + lm;                    // B col = W row (contig K)
            bf[j] = *(const v16bf*)(Wb + (size_t)col * D_ + k0 + kb16);
        }
        #pragma unroll
        for (int i = 0; i < 4; i++) {
            int row = m0 + i * 16 + lm;
            const float* ap = Ab + (size_t)row * D_ + k0;
            __builtin_prefetch(ap + 32, 0, 3);
            v4f a0 = *(const v4f*)(ap + kb16);
            v4f a1 = *(const v4f*)(ap + kb16 + 4);
            v4f a2 = *(const v4f*)(ap + kb16 + 8);
            v4f a3 = *(const v4f*)(ap + kb16 + 12);
            v16bf af;
            #pragma unroll
            for (int t = 0; t < 4; t++) {
                af[t]      = (__bf16)a0[t];
                af[4 + t]  = (__bf16)a1[t];
                af[8 + t]  = (__bf16)a2[t];
                af[12 + t] = (__bf16)a3[t];
            }
            #pragma unroll
            for (int j = 0; j < 4; j++)
                acc[i][j] = __builtin_amdgcn_wmma_f32_16x16x32_bf16(
                    false, af, false, bf[j], (short)0, acc[i][j], false, false);
        }
    }
    #pragma unroll
    for (int i = 0; i < 4; i++)
        #pragma unroll
        for (int j = 0; j < 4; j++) {
            int col = n0 + j * 16 + lm;
            #pragma unroll
            for (int t = 0; t < 8; t++) {
                int row = m0 + i * 16 + t + hi * 8;
                Cb[(size_t)row * D_ + col] = (__bf16)acc[i][j][t];
            }
        }
}

// ---------------------------------------------------------------------------
// GEMM 2: scores[b,q,s] = sum_d qb[b,q,d] * kWb[b,s,d]   (A bf16, B bf16, C f32)
// ---------------------------------------------------------------------------
__global__ __launch_bounds__(32)
void k_gemm_scores(const __bf16* __restrict__ qb, const __bf16* __restrict__ kWb,
                   float* __restrict__ scores) {
    const int lane = threadIdx.x;
    const int ntiles = S_ / 64, mtiles = Q_ / 64;
    int bid = blockIdx.x;
    int b  = bid / (mtiles * ntiles);
    int r  = bid % (mtiles * ntiles);
    int m0 = (r / ntiles) * 64;
    int n0 = (r % ntiles) * 64;

    const __bf16* Ab = qb  + (size_t)b * Q_ * D_;
    const __bf16* Bb = kWb + (size_t)b * S_ * D_;
    float*        Cb = scores + (size_t)b * Q_ * S_;

    const int lm   = lane & 15;
    const int hi   = lane >> 4;
    const int kb8  = hi * 8;
    const int kb16 = hi * 16;

    v8f acc[4][4];
    #pragma unroll
    for (int i = 0; i < 4; i++)
        #pragma unroll
        for (int j = 0; j < 4; j++) acc[i][j] = (v8f)0.0f;

    for (int k0 = 0; k0 < D_; k0 += 32) {
        v16bf bf[4];
        #pragma unroll
        for (int j = 0; j < 4; j++) {
            int col = n0 + j * 16 + lm;                    // B col = kW row (contig K)
            bf[j] = *(const v16bf*)(Bb + (size_t)col * D_ + k0 + kb16);
        }
        #pragma unroll
        for (int i = 0; i < 4; i++) {
            int row = m0 + i * 16 + lm;
            const __bf16* ap = Ab + (size_t)row * D_ + k0;
            __builtin_prefetch(ap + 32, 0, 3);
            v8bf c0 = *(const v8bf*)(ap + kb16);
            v8bf c1 = *(const v8bf*)(ap + kb16 + 8);
            v16bf af;
            #pragma unroll
            for (int t = 0; t < 8; t++) { af[t] = c0[t]; af[8 + t] = c1[t]; }
            #pragma unroll
            for (int j = 0; j < 4; j++)
                acc[i][j] = __builtin_amdgcn_wmma_f32_16x16x32_bf16(
                    false, af, false, bf[j], (short)0, acc[i][j], false, false);
        }
    }
    #pragma unroll
    for (int i = 0; i < 4; i++)
        #pragma unroll
        for (int j = 0; j < 4; j++) {
            int col = n0 + j * 16 + lm;
            #pragma unroll
            for (int t = 0; t < 8; t++) {
                int row = m0 + i * 16 + t + hi * 8;
                Cb[(size_t)row * S_ + col] = acc[i][j][t];
            }
        }
}

// ---------------------------------------------------------------------------
// Row softmax in place over attn [B*Q, S]; one 256-thread block per row.
// Optionally also emits a bf16 copy of the normalized row (attnb != nullptr
// is decided at compile time via template-like duplication below).
// ---------------------------------------------------------------------------
__device__ __forceinline__ void softmax_row(float* __restrict__ p,
                                            __bf16* __restrict__ pb,
                                            int tid, float* red, bool emit_bf16) {
    float x[16];
    float m = -3.402823e38f;
    #pragma unroll
    for (int i = 0; i < 16; i++) { x[i] = p[tid + i * 256]; m = fmaxf(m, x[i]); }
    #pragma unroll
    for (int off = 16; off > 0; off >>= 1) m = fmaxf(m, __shfl_xor(m, off, 32));
    if ((tid & 31) == 0) red[tid >> 5] = m;
    __syncthreads();
    float mAll = red[0];
    #pragma unroll
    for (int w = 1; w < 8; w++) mAll = fmaxf(mAll, red[w]);
    __syncthreads();

    float s = 0.0f;
    #pragma unroll
    for (int i = 0; i < 16; i++) { x[i] = __expf(x[i] - mAll); s += x[i]; }
    #pragma unroll
    for (int off = 16; off > 0; off >>= 1) s += __shfl_xor(s, off, 32);
    if ((tid & 31) == 0) red[tid >> 5] = s;
    __syncthreads();
    float sAll = 0.0f;
    #pragma unroll
    for (int w = 0; w < 8; w++) sAll += red[w];
    float inv = 1.0f / sAll;
    #pragma unroll
    for (int i = 0; i < 16; i++) {
        float y = x[i] * inv;
        p[tid + i * 256] = y;
        if (emit_bf16) pb[tid + i * 256] = (__bf16)y;
    }
}

__global__ __launch_bounds__(256)
void k_softmax(float* __restrict__ attn) {
    __shared__ float red[8];
    softmax_row(attn + (size_t)blockIdx.x * S_, nullptr, threadIdx.x, red, false);
}

__global__ __launch_bounds__(256)
void k_softmax_emit_bf16(float* __restrict__ attn, __bf16* __restrict__ attnb) {
    __shared__ float red[8];
    softmax_row(attn + (size_t)blockIdx.x * S_,
                attnb + (size_t)blockIdx.x * S_, threadIdx.x, red, true);
}

// ---------------------------------------------------------------------------
// GEMM 3 (fallback): out[b,q,v] = sum_s attn_f32[b,q,s] * valueT[b,v,s]
// ---------------------------------------------------------------------------
__global__ __launch_bounds__(32)
void k_gemm_out_f32A(const float* __restrict__ attn, const __bf16* __restrict__ vtb,
                     float* __restrict__ out) {
    const int lane = threadIdx.x;
    const int ntiles = D_ / 64, mtiles = Q_ / 64;
    int bid = blockIdx.x;
    int b  = bid / (mtiles * ntiles);
    int r  = bid % (mtiles * ntiles);
    int m0 = (r / ntiles) * 64;
    int n0 = (r % ntiles) * 64;

    const float*  Ab = attn + (size_t)b * Q_ * S_;
    const __bf16* Bb = vtb  + (size_t)b * D_ * S_;
    float*        Cb = out  + (size_t)b * Q_ * D_;

    const int lm   = lane & 15;
    const int hi   = lane >> 4;
    const int kb8  = hi * 8;
    const int kb16 = hi * 16;

    v8f acc[4][4];
    #pragma unroll
    for (int i = 0; i < 4; i++)
        #pragma unroll
        for (int j = 0; j < 4; j++) acc[i][j] = (v8f)0.0f;

    for (int k0 = 0; k0 < S_; k0 += 32) {
        v16bf bf[4];
        #pragma unroll
        for (int j = 0; j < 4; j++) {
            int col = n0 + j * 16 + lm;                    // B col = valueT row (contig K=s)
            bf[j] = *(const v16bf*)(Bb + (size_t)col * S_ + k0 + kb16);
        }
        #pragma unroll
        for (int i = 0; i < 4; i++) {
            int row = m0 + i * 16 + lm;
            const float* ap = Ab + (size_t)row * S_ + k0;
            __builtin_prefetch(ap + 32, 0, 3);
            v4f a0 = *(const v4f*)(ap + kb16);
            v4f a1 = *(const v4f*)(ap + kb16 + 4);
            v4f a2 = *(const v4f*)(ap + kb16 + 8);
            v4f a3 = *(const v4f*)(ap + kb16 + 12);
            v16bf af;
            #pragma unroll
            for (int t = 0; t < 4; t++) {
                af[t]      = (__bf16)a0[t];
                af[4 + t]  = (__bf16)a1[t];
                af[8 + t]  = (__bf16)a2[t];
                af[12 + t] = (__bf16)a3[t];
            }
            #pragma unroll
            for (int j = 0; j < 4; j++)
                acc[i][j] = __builtin_amdgcn_wmma_f32_16x16x32_bf16(
                    false, af, false, bf[j], (short)0, acc[i][j], false, false);
        }
    }
    #pragma unroll
    for (int i = 0; i < 4; i++)
        #pragma unroll
        for (int j = 0; j < 4; j++) {
            int col = n0 + j * 16 + lm;
            #pragma unroll
            for (int t = 0; t < 8; t++) {
                int row = m0 + i * 16 + t + hi * 8;
                Cb[(size_t)row * D_ + col] = acc[i][j][t];
            }
        }
}

// ---------------------------------------------------------------------------
// GEMM 3 (fast path): out[b,q,v] = sum_s attn_bf16[b,q,s] * valueT[b,v,s]
// Pure bf16 loads on both operands; no conversions in the hot loop.
// ---------------------------------------------------------------------------
__global__ __launch_bounds__(32)
void k_gemm_out_bf16A(const __bf16* __restrict__ attnb, const __bf16* __restrict__ vtb,
                      float* __restrict__ out) {
    const int lane = threadIdx.x;
    const int ntiles = D_ / 64, mtiles = Q_ / 64;
    int bid = blockIdx.x;
    int b  = bid / (mtiles * ntiles);
    int r  = bid % (mtiles * ntiles);
    int m0 = (r / ntiles) * 64;
    int n0 = (r % ntiles) * 64;

    const __bf16* Ab = attnb + (size_t)b * Q_ * S_;
    const __bf16* Bb = vtb   + (size_t)b * D_ * S_;
    float*        Cb = out   + (size_t)b * Q_ * D_;

    const int lm   = lane & 15;
    const int hi   = lane >> 4;
    const int kb8  = hi * 8;
    const int kb16 = hi * 16;

    v8f acc[4][4];
    #pragma unroll
    for (int i = 0; i < 4; i++)
        #pragma unroll
        for (int j = 0; j < 4; j++) acc[i][j] = (v8f)0.0f;

    for (int k0 = 0; k0 < S_; k0 += 32) {
        v16bf bf[4];
        #pragma unroll
        for (int j = 0; j < 4; j++) {
            int col = n0 + j * 16 + lm;
            bf[j] = *(const v16bf*)(Bb + (size_t)col * S_ + k0 + kb16);
        }
        #pragma unroll
        for (int i = 0; i < 4; i++) {
            int row = m0 + i * 16 + lm;
            const __bf16* ap = Ab + (size_t)row * S_ + k0;
            __builtin_prefetch(ap + 32, 0, 3);
            v8bf c0 = *(const v8bf*)(ap + kb16);
            v8bf c1 = *(const v8bf*)(ap + kb16 + 8);
            v16bf af;
            #pragma unroll
            for (int t = 0; t < 8; t++) { af[t] = c0[t]; af[8 + t] = c1[t]; }
            #pragma unroll
            for (int j = 0; j < 4; j++)
                acc[i][j] = __builtin_amdgcn_wmma_f32_16x16x32_bf16(
                    false, af, false, bf[j], (short)0, acc[i][j], false, false);
        }
    }
    #pragma unroll
    for (int i = 0; i < 4; i++)
        #pragma unroll
        for (int j = 0; j < 4; j++) {
            int col = n0 + j * 16 + lm;
            #pragma unroll
            for (int t = 0; t < 8; t++) {
                int row = m0 + i * 16 + t + hi * 8;
                Cb[(size_t)row * D_ + col] = acc[i][j][t];
            }
        }
}

// ---------------------------------------------------------------------------
extern "C" void kernel_launch(void* const* d_in, const int* in_sizes, int n_in,
                              void* d_out, int out_size, void* d_ws, size_t ws_size,
                              hipStream_t stream) {
    const float* value = (const float*)d_in[0];
    const float* key   = (const float*)d_in[1];
    const float* query = (const float*)d_in[2];
    const float* W     = (const float*)d_in[3];

    float* out  = (float*)d_out;                       // [B,Q,D]
    float* attn = out + (size_t)B_ * Q_ * D_;          // [B,Q,S]

    // workspace layout (bf16):
    __bf16* kWb   = (__bf16*)d_ws;                     // [B,S,D]  32 MiB
    __bf16* qb    = kWb + (size_t)B_ * S_ * D_;        // [B,Q,D]  32 MiB
    __bf16* vtb   = qb  + (size_t)B_ * Q_ * D_;        // [B,D,S]  32 MiB
    __bf16* Wb    = vtb + (size_t)B_ * S_ * D_;        // [D,D]     2 MiB
    __bf16* attnb = Wb  + (size_t)D_ * D_;             // [B,Q,S]  128 MiB (optional)

    const size_t base_elems =
        (size_t)B_ * S_ * D_ + (size_t)B_ * Q_ * D_ + (size_t)B_ * S_ * D_ +
        (size_t)D_ * D_;
    const size_t need_bf16_attn =
        (base_elems + (size_t)B_ * Q_ * S_) * sizeof(__bf16);
    const bool use_bf16_attn = (ws_size >= need_bf16_attn);

    k_pack_bf16  <<<(D_ * D_) / 1024,                  256, 0, stream>>>(W, Wb, D_ * D_);
    k_pack_bf16  <<<((size_t)B_ * Q_ * D_) / 1024,     256, 0, stream>>>(query, qb, B_ * Q_ * D_);
    k_pack_valueT<<<B_ * (S_ / 32) * (D_ / 32),        256, 0, stream>>>(value, vtb);

    k_gemm_kW    <<<B_ * (S_ / 64) * (D_ / 64),         32, 0, stream>>>(key, Wb, kWb);
    k_gemm_scores<<<B_ * (Q_ / 64) * (S_ / 64),         32, 0, stream>>>(qb, kWb, attn);

    if (use_bf16_attn) {
        k_softmax_emit_bf16<<<B_ * Q_,                 256, 0, stream>>>(attn, attnb);
        k_gemm_out_bf16A   <<<B_ * (Q_ / 64) * (D_ / 64), 32, 0, stream>>>(attnb, vtb, out);
    } else {
        k_softmax          <<<B_ * Q_,                 256, 0, stream>>>(attn);
        k_gemm_out_f32A    <<<B_ * (Q_ / 64) * (D_ / 64), 32, 0, stream>>>(attn, vtb, out);
    }
}
